// RelationalLayerModule_21973052686562
// MI455X (gfx1250) — matrix operand order repaired and checked by reading, not compiled, not run on hardware
//
#include <hip/hip_runtime.h>

typedef __attribute__((ext_vector_type(16))) _Float16 v16h;
typedef __attribute__((ext_vector_type(8)))  _Float16 v8h;
typedef __attribute__((ext_vector_type(8)))  float    v8f;

#define EMB 64

// ---------------------------------------------------------------------------
// Fragment loaders (CDNA5 WMMA 16x16x32 f16 layouts, wave32)
// ---------------------------------------------------------------------------

// A-matrix 16x32 f16: lane L holds row M=L&15; lanes<16 hold K {kb..kb+7, kb+16..kb+23},
// lanes>=16 hold K {kb+8..kb+15, kb+24..kb+31}. Two contiguous 16B chunks per lane.
__device__ __forceinline__ v16h load_a_frag(const _Float16* tile, int rs, int kb, int lane) {
  int r  = lane & 15;
  int k0 = kb + ((lane >> 4) << 3);
  const _Float16* p = tile + r * rs + k0;
  v8h lo = *(const v8h*)(p);
  v8h hi = *(const v8h*)(p + 16);
  v16h a;
#pragma unroll
  for (int i = 0; i < 8; ++i) { a[i] = lo[i]; a[i + 8] = hi[i]; }
  return a;
}

// B-matrix 32x16 f16: lane L holds col N=n0+(L&15); lanes<16 hold K {kb..kb+7, kb+16..kb+23},
// lanes>=16 hold K {kb+8..kb+15, kb+24..kb+31}. Wt is N-major (Wt[n][k]): two 16B chunks.
__device__ __forceinline__ v16h load_b_frag(const _Float16* __restrict__ Wt, int K,
                                            int n0, int kb, int lane) {
  int c  = n0 + (lane & 15);
  int k0 = kb + ((lane >> 4) << 3);
  const _Float16* p = Wt + (long)c * K + k0;
  v8h lo = *(const v8h*)(p);
  v8h hi = *(const v8h*)(p + 16);
  v16h b;
#pragma unroll
  for (int i = 0; i < 8; ++i) { b[i] = lo[i]; b[i + 8] = hi[i]; }
  return b;
}

__device__ __forceinline__ v8f wmma_f16(v16h a, v16h b, v8f c) {
  return __builtin_amdgcn_wmma_f32_16x16x32_f16(false, a, false, b, (short)0, c, false, false);
}

// Convert NF f32 from global to f16 in LDS (NF multiple of 8, 16B-aligned dst).
template <int NF>
__device__ __forceinline__ void cvt_store_row(const float* __restrict__ src, _Float16* dst) {
  const float4* s4 = (const float4*)src;
#pragma unroll
  for (int i = 0; i < NF / 8; ++i) {
    float4 u = s4[2 * i], v = s4[2 * i + 1];
    v8h h;
    h[0] = (_Float16)u.x; h[1] = (_Float16)u.y; h[2] = (_Float16)u.z; h[3] = (_Float16)u.w;
    h[4] = (_Float16)v.x; h[5] = (_Float16)v.y; h[6] = (_Float16)v.z; h[7] = (_Float16)v.w;
    *(v8h*)(dst + 8 * i) = h;
  }
}

// ---------------------------------------------------------------------------
// Prep kernels
// ---------------------------------------------------------------------------
__global__ void zero_f4(float4* __restrict__ p, long n4) {
  long i = (long)blockIdx.x * blockDim.x + threadIdx.x;
  if (i < n4) p[i] = make_float4(0.f, 0.f, 0.f, 0.f);
}

// Wt[n*K + k] = (f16) W[k*N + n]
__global__ void transpose_to_f16(const float* __restrict__ W, _Float16* __restrict__ Wt,
                                 int K, int N) {
  int i = blockIdx.x * blockDim.x + threadIdx.x;
  if (i >= K * N) return;
  int k = i / N, n = i - k * N;
  Wt[(long)n * K + k] = (_Float16)W[i];
}

// ---------------------------------------------------------------------------
// Relation 1 (arity 1): msg = MLP64(emb[idx]); atomic scatter into agg[idx].
// Wave = 32 atoms (two 16-row WMMA tiles sharing every B fragment).
// Hidden tile aliases input tile (input fully register-resident first).
// ---------------------------------------------------------------------------
__global__ void __launch_bounds__(256) rel1_msgs(
    const float* __restrict__ emb, const int* __restrict__ idx,
    const _Float16* __restrict__ W1t, const float* __restrict__ b1,
    const _Float16* __restrict__ W2t, const float* __restrict__ b2,
    float* __restrict__ agg, int M) {
  __shared__ __attribute__((aligned(16))) _Float16 gt[8][32][64];  // 32 KB
  __shared__ int ids[8][32];
  const int lane = threadIdx.x & 31;
  const int wv   = threadIdx.x >> 5;
  const long base = (long)blockIdx.x * 256 + (long)wv * 32;

  { // gather: one full 64-float row per lane
    long a = base + lane;
    int nid = (a < M) ? idx[a] : 0;
    ids[wv][lane] = nid;
    cvt_store_row<64>(emb + (long)nid * EMB, &gt[wv][lane][0]);
  }
  __syncthreads();

  const int col16 = lane & 15;
  const int rbase = (lane >> 4) << 3;

  // Layer 1: entire 32x64 input tile into registers, then overwrite with hidden.
  v16h af[2][2];
#pragma unroll
  for (int t = 0; t < 2; ++t)
#pragma unroll
    for (int k = 0; k < 2; ++k)
      af[t][k] = load_a_frag(&gt[wv][t * 16][0], 64, k * 32, lane);

#pragma unroll
  for (int nt = 0; nt < 4; ++nt) {
    v8f acc0 = {}, acc1 = {};
#pragma unroll
    for (int k = 0; k < 2; ++k) {
      v16h b = load_b_frag(W1t, 64, nt * 16, k * 32, lane);
      acc0 = wmma_f16(af[0][k], b, acc0);
      acc1 = wmma_f16(af[1][k], b, acc1);
    }
    int col = nt * 16 + col16;
    float bb = b1[col];
#pragma unroll
    for (int j = 0; j < 8; ++j) {
      float v0 = acc0[j] + bb, v1 = acc1[j] + bb;
      gt[wv][rbase + j][col]      = (_Float16)(v0 > 0.f ? v0 : 0.f);
      gt[wv][16 + rbase + j][col] = (_Float16)(v1 > 0.f ? v1 : 0.f);
    }
  }

  // Layer 2 + atomic scatter.
  v16h hf[2][2];
#pragma unroll
  for (int t = 0; t < 2; ++t)
#pragma unroll
    for (int k = 0; k < 2; ++k)
      hf[t][k] = load_a_frag(&gt[wv][t * 16][0], 64, k * 32, lane);

#pragma unroll
  for (int nt = 0; nt < 4; ++nt) {
    v8f acc0 = {}, acc1 = {};
#pragma unroll
    for (int k = 0; k < 2; ++k) {
      v16h b = load_b_frag(W2t, 64, nt * 16, k * 32, lane);
      acc0 = wmma_f16(hf[0][k], b, acc0);
      acc1 = wmma_f16(hf[1][k], b, acc1);
    }
    int col = nt * 16 + col16;
    float bb = b2[col];
#pragma unroll
    for (int j = 0; j < 8; ++j) {
      int r0 = rbase + j, r1 = 16 + rbase + j;
      if (base + r0 < M)
        atomicAdd(&agg[(long)ids[wv][r0] * EMB + col], acc0[j] + bb);
      if (base + r1 < M)
        atomicAdd(&agg[(long)ids[wv][r1] * EMB + col], acc1[j] + bb);
    }
  }
}

// ---------------------------------------------------------------------------
// Relation 2 (arity 2): g=[x_a|x_b] (128); MLP128; split message scatter.
// Wave = 32 atoms; 128 WMMAs per wave; B fragments shared across both tiles.
// ---------------------------------------------------------------------------
__global__ void __launch_bounds__(256) rel2_msgs(
    const float* __restrict__ emb, const int* __restrict__ idx,
    const _Float16* __restrict__ W1t, const float* __restrict__ b1,
    const _Float16* __restrict__ W2t, const float* __restrict__ b2,
    float* __restrict__ agg, int M) {
  __shared__ __attribute__((aligned(16))) _Float16 gt[8][32][128]; // 64 KB
  __shared__ int ids[8][32][2];
  const int lane = threadIdx.x & 31;
  const int wv   = threadIdx.x >> 5;
  const long base = (long)blockIdx.x * 256 + (long)wv * 32;

  { // gather: each lane pulls both argument rows for its atom
    long a = base + lane;
    int n0 = (a < M) ? idx[2 * a]     : 0;
    int n1 = (a < M) ? idx[2 * a + 1] : 0;
    ids[wv][lane][0] = n0;
    ids[wv][lane][1] = n1;
    cvt_store_row<64>(emb + (long)n0 * EMB, &gt[wv][lane][0]);
    cvt_store_row<64>(emb + (long)n1 * EMB, &gt[wv][lane][64]);
  }
  __syncthreads();

  const int col16 = lane & 15;
  const int rbase = (lane >> 4) << 3;

  // Layer 1: full 32x128 input tile into registers (64 VGPRs), alias hidden onto it.
  v16h af[2][4];
#pragma unroll
  for (int t = 0; t < 2; ++t)
#pragma unroll
    for (int k = 0; k < 4; ++k)
      af[t][k] = load_a_frag(&gt[wv][t * 16][0], 128, k * 32, lane);

#pragma unroll
  for (int nt = 0; nt < 8; ++nt) {
    v8f acc0 = {}, acc1 = {};
#pragma unroll
    for (int k = 0; k < 4; ++k) {
      v16h b = load_b_frag(W1t, 128, nt * 16, k * 32, lane);
      acc0 = wmma_f16(af[0][k], b, acc0);
      acc1 = wmma_f16(af[1][k], b, acc1);
    }
    int col = nt * 16 + col16;
    float bb = b1[col];
#pragma unroll
    for (int j = 0; j < 8; ++j) {
      float v0 = acc0[j] + bb, v1 = acc1[j] + bb;
      gt[wv][rbase + j][col]      = (_Float16)(v0 > 0.f ? v0 : 0.f);
      gt[wv][16 + rbase + j][col] = (_Float16)(v1 > 0.f ? v1 : 0.f);
    }
  }

  // Layer 2 + split atomic scatter (cols 0..63 -> arg0, 64..127 -> arg1).
  v16h hf[2][4];
#pragma unroll
  for (int t = 0; t < 2; ++t)
#pragma unroll
    for (int k = 0; k < 4; ++k)
      hf[t][k] = load_a_frag(&gt[wv][t * 16][0], 128, k * 32, lane);

#pragma unroll
  for (int nt = 0; nt < 8; ++nt) {
    v8f acc0 = {}, acc1 = {};
#pragma unroll
    for (int k = 0; k < 4; ++k) {
      v16h b = load_b_frag(W2t, 128, nt * 16, k * 32, lane);
      acc0 = wmma_f16(hf[0][k], b, acc0);
      acc1 = wmma_f16(hf[1][k], b, acc1);
    }
    int col = nt * 16 + col16;
    float bb = b2[col];
    int arg = col >> 6, feat = col & 63;
#pragma unroll
    for (int j = 0; j < 8; ++j) {
      int r0 = rbase + j, r1 = 16 + rbase + j;
      if (base + r0 < M)
        atomicAdd(&agg[(long)ids[wv][r0][arg] * EMB + feat], acc0[j] + bb);
      if (base + r1 < M)
        atomicAdd(&agg[(long)ids[wv][r1][arg] * EMB + feat], acc1[j] + bb);
    }
  }
}

// ---------------------------------------------------------------------------
// Node update: u=[emb|agg] (128); relu(u@W1+b1)@W2+b2 -> out. No atomics.
// Wave = 32 nodes; hidden (32x64) aliases first half of each input row.
// ---------------------------------------------------------------------------
__global__ void __launch_bounds__(256) node_update(
    const float* __restrict__ emb, const float* __restrict__ agg,
    const _Float16* __restrict__ W1t, const float* __restrict__ b1,
    const _Float16* __restrict__ W2t, const float* __restrict__ b2,
    float* __restrict__ out, int NN) {
  __shared__ __attribute__((aligned(16))) _Float16 gt[8][32][128]; // 64 KB
  const int lane = threadIdx.x & 31;
  const int wv   = threadIdx.x >> 5;
  const long base = (long)blockIdx.x * 256 + (long)wv * 32;

  { // gather: lane's node -> [emb row | agg row]
    long n  = base + lane;
    long nn = (n < NN) ? n : 0;
    cvt_store_row<64>(emb + nn * EMB, &gt[wv][lane][0]);
    cvt_store_row<64>(agg + nn * EMB, &gt[wv][lane][64]);
  }
  __syncthreads();

  const int col16 = lane & 15;
  const int rbase = (lane >> 4) << 3;

  v16h af[2][4];
#pragma unroll
  for (int t = 0; t < 2; ++t)
#pragma unroll
    for (int k = 0; k < 4; ++k)
      af[t][k] = load_a_frag(&gt[wv][t * 16][0], 128, k * 32, lane);

#pragma unroll
  for (int nt = 0; nt < 4; ++nt) {
    v8f acc0 = {}, acc1 = {};
#pragma unroll
    for (int k = 0; k < 4; ++k) {
      v16h b = load_b_frag(W1t, 128, nt * 16, k * 32, lane);
      acc0 = wmma_f16(af[0][k], b, acc0);
      acc1 = wmma_f16(af[1][k], b, acc1);
    }
    int col = nt * 16 + col16;
    float bb = b1[col];
#pragma unroll
    for (int j = 0; j < 8; ++j) {
      float v0 = acc0[j] + bb, v1 = acc1[j] + bb;
      gt[wv][rbase + j][col]      = (_Float16)(v0 > 0.f ? v0 : 0.f);
      gt[wv][16 + rbase + j][col] = (_Float16)(v1 > 0.f ? v1 : 0.f);
    }
  }

  // Layer 2: hidden rows live in cols 0..63 of each 128-half row (stride 128).
  v16h hf[2][2];
#pragma unroll
  for (int t = 0; t < 2; ++t)
#pragma unroll
    for (int k = 0; k < 2; ++k)
      hf[t][k] = load_a_frag(&gt[wv][t * 16][0], 128, k * 32, lane);

#pragma unroll
  for (int nt = 0; nt < 4; ++nt) {
    v8f acc0 = {}, acc1 = {};
#pragma unroll
    for (int k = 0; k < 2; ++k) {
      v16h b = load_b_frag(W2t, 64, nt * 16, k * 32, lane);
      acc0 = wmma_f16(hf[0][k], b, acc0);
      acc1 = wmma_f16(hf[1][k], b, acc1);
    }
    int col = nt * 16 + col16;
    float bb = b2[col];
#pragma unroll
    for (int j = 0; j < 8; ++j) {
      long n0 = base + rbase + j, n1 = base + 16 + rbase + j;
      if (n0 < NN) out[n0 * EMB + col] = acc0[j] + bb;
      if (n1 < NN) out[n1 * EMB + col] = acc1[j] + bb;
    }
  }
}

// ---------------------------------------------------------------------------
// Host launcher
// ---------------------------------------------------------------------------
extern "C" void kernel_launch(void* const* d_in, const int* in_sizes, int n_in,
                              void* d_out, int out_size, void* d_ws, size_t ws_size,
                              hipStream_t stream) {
  const float* emb  = (const float*)d_in[0];
  const int*   r1   = (const int*)d_in[1];
  const int*   r2   = (const int*)d_in[2];
  const float* m1W1 = (const float*)d_in[3];
  const float* m1b1 = (const float*)d_in[4];
  const float* m1W2 = (const float*)d_in[5];
  const float* m1b2 = (const float*)d_in[6];
  const float* m2W1 = (const float*)d_in[7];
  const float* m2b1 = (const float*)d_in[8];
  const float* m2W2 = (const float*)d_in[9];
  const float* m2b2 = (const float*)d_in[10];
  const float* uW1  = (const float*)d_in[11];
  const float* ub1  = (const float*)d_in[12];
  const float* uW2  = (const float*)d_in[13];
  const float* ub2  = (const float*)d_in[14];

  const int NN = in_sizes[0] / EMB;  // 100000
  const int M1 = in_sizes[1];        // 500000
  const int M2 = in_sizes[2] / 2;    // 1000000

  char* ws = (char*)d_ws;
  float* agg = (float*)ws;
  size_t off = (size_t)NN * EMB * sizeof(float);
  off = (off + 255) & ~(size_t)255;
  _Float16* w1m1t = (_Float16*)(ws + off); off += 64 * 64 * 2;
  _Float16* w2m1t = (_Float16*)(ws + off); off += 64 * 64 * 2;
  _Float16* w1m2t = (_Float16*)(ws + off); off += 128 * 128 * 2;
  _Float16* w2m2t = (_Float16*)(ws + off); off += 128 * 128 * 2;
  _Float16* w1ut  = (_Float16*)(ws + off); off += 128 * 64 * 2;
  _Float16* w2ut  = (_Float16*)(ws + off); off += 64 * 64 * 2;

  // Zero the aggregation buffer every call (deterministic, atomics target).
  long n4 = (long)NN * EMB / 4;
  zero_f4<<<dim3((unsigned)((n4 + 255) / 256)), dim3(256), 0, stream>>>((float4*)agg, n4);

  // Convert + transpose weights to f16 N-major once per call (tiny).
  auto tr = [&](const float* W, _Float16* Wt, int K, int Nc) {
    int tot = K * Nc;
    transpose_to_f16<<<dim3((tot + 255) / 256), dim3(256), 0, stream>>>(W, Wt, K, Nc);
  };
  tr(m1W1, w1m1t, 64, 64);
  tr(m1W2, w2m1t, 64, 64);
  tr(m2W1, w1m2t, 128, 128);
  tr(m2W2, w2m2t, 128, 128);
  tr(uW1,  w1ut, 128, 64);
  tr(uW2,  w2ut, 64, 64);

  rel1_msgs<<<dim3((M1 + 255) / 256), dim3(256), 0, stream>>>(
      emb, r1, w1m1t, m1b1, w2m1t, m1b2, agg, M1);
  rel2_msgs<<<dim3((M2 + 255) / 256), dim3(256), 0, stream>>>(
      emb, r2, w1m2t, m2b1, w2m2t, m2b2, agg, M2);
  node_update<<<dim3((NN + 255) / 256), dim3(256), 0, stream>>>(
      emb, agg, w1ut, ub1, w2ut, ub2, (float*)d_out, NN);
}
